// GIN_18107582120449
// MI455X (gfx1250) — compile-verified, matrix operand order checked by reading, not run on hardware
//
#include <hip/hip_runtime.h>
#include <hip/hip_bf16.h>
#include <math.h>

typedef __attribute__((ext_vector_type(16))) _Float16 v16h;
typedef __attribute__((ext_vector_type(8)))  _Float16 v8h;
typedef __attribute__((ext_vector_type(4)))  _Float16 v4h;
typedef __attribute__((ext_vector_type(8)))  float    v8f;

#define N_NODES  50000
#define N_EDGES  400000
#define D_IN     512
#define DIM_H    256
#define N_GRAPHS 64
#define BN_EPS   1e-5f

// ---------------- utility kernels ----------------

__global__ void k_copy4(const float* __restrict__ in, float* __restrict__ out, int n4) {
  int i = blockIdx.x * blockDim.x + threadIdx.x;
  if (i < n4) ((float4*)out)[i] = ((const float4*)in)[i];
}

__global__ void k_zero(float* __restrict__ p, int n) {
  int i = blockIdx.x * blockDim.x + threadIdx.x;
  if (i < n) p[i] = 0.f;
}

// W[K x N] f32 (row-major, JAX layout)  ->  Bt[N x K] f16 (N-major / transposed)
__global__ void k_convert_wT(const float* __restrict__ in, _Float16* __restrict__ out,
                             int K, int N) {
  int i = blockIdx.x * blockDim.x + threadIdx.x;
  if (i >= K * N) return;
  int n = i / K, k = i - n * K;
  out[i] = (_Float16)in[(size_t)k * N + n];
}

// agg[dst] += x[src] over edges; agg pre-initialized to x (GIN eps=0).
__global__ void k_scatter_add(const float* __restrict__ x, const int* __restrict__ src,
                              const int* __restrict__ dst, float* __restrict__ agg, int D) {
  int i = blockIdx.x * blockDim.x + threadIdx.x;
  int per = D >> 2;
  int e = i / per;
  if (e >= N_EDGES) return;
  int c = (i - e * per) << 2;
  const float4 v = *(const float4*)(x + (size_t)src[e] * D + c);
  float* a = agg + (size_t)dst[e] * D + c;
  atomicAdd(a + 0, v.x); atomicAdd(a + 1, v.y);
  atomicAdd(a + 2, v.z); atomicAdd(a + 3, v.w);
}

// global_add_pool: pooled[batch[n]][colOff + c] += h[n][c]   (pooled ld = 3*DIM_H)
__global__ void k_pool(const float* __restrict__ h, const int* __restrict__ batch,
                       float* __restrict__ pooled, int colOff) {
  int i = blockIdx.x * blockDim.x + threadIdx.x;
  int per = DIM_H >> 2;
  int node = i / per;
  if (node >= N_NODES) return;
  int c = (i - node * per) << 2;
  int g = batch[node];
  const float4 v = *(const float4*)(h + (size_t)node * DIM_H + c);
  float* p = pooled + (size_t)g * (3 * DIM_H) + colOff + c;
  atomicAdd(p + 0, v.x); atomicAdd(p + 1, v.y);
  atomicAdd(p + 2, v.z); atomicAdd(p + 3, v.w);
}

// ---------------- WMMA GEMM: C = epilogue(A[f32,MxK] @ Bt[f16,NxK]^T) ----------------
// Block tile 256x64, K-step 64; 8 waves, each wave -> 32(M) x 64(N) strip (acc[2][4]),
// so each B fragment is reused across 2 A fragments (1.5 ds_load_b128 per WMMA).
// Staging: load phase fully issued before convert/store phase (overlapped loadcnt).
__global__ __launch_bounds__(256) void k_wmma_gemm(
    const float* __restrict__ A, const _Float16* __restrict__ Bt, float* __restrict__ C,
    int M, int N, int K,
    const float* __restrict__ bias,
    const float* __restrict__ gamma, const float* __restrict__ beta,
    const float* __restrict__ mean,  const float* __restrict__ var,
    int relu)
{
  __shared__ __align__(16) _Float16 As[256][72];   // 256 x 64, padded (144B row stride)
  __shared__ __align__(16) _Float16 BsT[64][72];   // 64(N) x 64(K), padded

  const int tid  = threadIdx.x;
  const int lane = tid & 31;
  const int wv   = tid >> 5;            // wave id 0..7 -> 32-row M strip
  const int m0   = blockIdx.y * 256;
  const int n0   = blockIdx.x * 64;
  const int ml   = lane & 15;           // row (A) / col (B) within 16
  const int kb   = (lane >> 4) << 3;    // K sub-base per ISA layout: 0 or 8

  v8f acc[2][4] = {};

  for (int k0 = 0; k0 < K; k0 += 64) {
    __syncthreads();
    // stage A tile 256x64 (f32 -> f16): 16 float4 per thread in 2 batches;
    // loads issued first, then converted/stored (no per-load waitcnt-0 stalls).
#pragma unroll
    for (int b = 0; b < 2; ++b) {
      float4 va[8];
#pragma unroll
      for (int j = 0; j < 8; ++j) {
        int li = b * 2048 + j * 256 + tid;
        int r = li >> 4, c4 = (li & 15) << 2;
        int gm  = m0 + r;
        int gmc = (gm < M) ? gm : (M - 1);
        va[j] = *(const float4*)(A + (size_t)gmc * K + (k0 + c4));
      }
#pragma unroll
      for (int j = 0; j < 8; ++j) {
        int li = b * 2048 + j * 256 + tid;
        int r = li >> 4, c4 = (li & 15) << 2;
        float4 v = va[j];
        if (m0 + r >= M) { v.x = 0.f; v.y = 0.f; v.z = 0.f; v.w = 0.f; }
        v4h h = { (_Float16)v.x, (_Float16)v.y, (_Float16)v.z, (_Float16)v.w };
        *(v4h*)&As[r][c4] = h;
      }
    }
    // stage B tile 64(N) x 64(K): 2 coalesced 16B copies per thread
#pragma unroll
    for (int j = 0; j < 2; ++j) {
      int li = j * 256 + tid;
      int n = li >> 3, kc = (li & 7) << 3;
      *(v8h*)&BsT[n][kc] = *(const v8h*)(Bt + (size_t)(n0 + n) * K + (k0 + kc));
    }
    // prefetch next A tile through GL2 while we compute
    if (k0 + 64 < K) {
      int pr = m0 + tid;
      if (pr < M) __builtin_prefetch(&A[(size_t)pr * K + (k0 + 64)], 0, 3);
    }
    __syncthreads();

    // two 16x16x32 K-subtiles per stage, 2 M-subtiles x 4 N-subtiles per wave
#pragma unroll
    for (int s = 0; s < 64; s += 32) {
      v16h af[2];
#pragma unroll
      for (int u = 0; u < 2; ++u) {
        // A fragment: lanes 0-15 row K{0..7,16..23}; lanes 16-31 K{8..15,24..31}
        v8h alo = *(const v8h*)&As[wv * 32 + u * 16 + ml][s + kb];
        v8h ahi = *(const v8h*)&As[wv * 32 + u * 16 + ml][s + kb + 16];
        af[u] = __builtin_shufflevector(alo, ahi, 0,1,2,3,4,5,6,7,8,9,10,11,12,13,14,15);
      }
#pragma unroll
      for (int t = 0; t < 4; ++t) {
        v8h blo = *(const v8h*)&BsT[t * 16 + ml][s + kb];
        v8h bhi = *(const v8h*)&BsT[t * 16 + ml][s + kb + 16];
        v16h bf = __builtin_shufflevector(blo, bhi, 0,1,2,3,4,5,6,7,8,9,10,11,12,13,14,15);
#pragma unroll
        for (int u = 0; u < 2; ++u) {
          acc[u][t] = __builtin_amdgcn_wmma_f32_16x16x32_f16(
              /*neg_a=*/false, af[u], /*neg_b=*/false, bf,
              /*c_mod=*/(short)0, acc[u][t], /*reuse_a=*/false, /*reuse_b=*/false);
        }
      }
    }
  }

  // epilogue: C/D layout — VGPR r: M = r + 8*(lane>=16), N = lane&15
  const int rb = (lane >> 4) << 3;
#pragma unroll
  for (int t = 0; t < 4; ++t) {
    int n = n0 + t * 16 + ml;
    float sc = 1.f, sh = bias[n];
    if (gamma) {
      float is = rsqrtf(var[n] + BN_EPS) * gamma[n];
      sc = is;
      sh = (bias[n] - mean[n]) * is + beta[n];
    }
#pragma unroll
    for (int u = 0; u < 2; ++u) {
#pragma unroll
      for (int r = 0; r < 8; ++r) {
        int m = m0 + wv * 32 + u * 16 + rb + r;
        if (m < M) {
          float v = acc[u][t][r] * sc + sh;
          if (relu) v = fmaxf(v, 0.f);
          C[(size_t)m * N + n] = v;
        }
      }
    }
  }
}

// ---------------- head: lin2 + log_softmax over 2 classes ----------------
__global__ void k_head(const float* __restrict__ lin1o, const float* __restrict__ W2,
                       const float* __restrict__ b2, float* __restrict__ out) {
  int r = threadIdx.x;
  if (r >= N_GRAPHS) return;
  float l0 = b2[0], l1 = b2[1];
#pragma unroll 8
  for (int k = 0; k < 64; ++k) {
    float a = lin1o[r * 64 + k];
    l0 += a * W2[k * 2 + 0];
    l1 += a * W2[k * 2 + 1];
  }
  out[r * 2 + 0] = l0;
  out[r * 2 + 1] = l1;
  float mx = fmaxf(l0, l1);
  float lse = mx + logf(expf(l0 - mx) + expf(l1 - mx));
  out[2 * N_GRAPHS + r * 2 + 0] = l0 - lse;
  out[2 * N_GRAPHS + r * 2 + 1] = l1 - lse;
}

// ---------------- launcher ----------------
extern "C" void kernel_launch(void* const* d_in, const int* in_sizes, int n_in,
                              void* d_out, int out_size, void* d_ws, size_t ws_size,
                              hipStream_t stream) {
  (void)in_sizes; (void)n_in; (void)out_size; (void)ws_size;

  const float* x     = (const float*)d_in[0];
  const int*   ei    = (const int*)d_in[1];
  const int*   src   = ei;
  const int*   dst   = ei + N_EDGES;
  const int*   batch = (const int*)d_in[2];

  const float* W1[3], *b1[3], *ga[3], *be[3], *mu[3], *va[3], *W2[3], *b2[3];
  for (int l = 0; l < 3; ++l) {
    int base = 3 + 8 * l;
    W1[l] = (const float*)d_in[base + 0];  b1[l] = (const float*)d_in[base + 1];
    ga[l] = (const float*)d_in[base + 2];  be[l] = (const float*)d_in[base + 3];
    mu[l] = (const float*)d_in[base + 4];  va[l] = (const float*)d_in[base + 5];
    W2[l] = (const float*)d_in[base + 6];  b2[l] = (const float*)d_in[base + 7];
  }
  const float* lin1W = (const float*)d_in[27];
  const float* lin1b = (const float*)d_in[28];
  const float* lin2W = (const float*)d_in[29];
  const float* lin2b = (const float*)d_in[30];

  // workspace layout (all region sizes are multiples of 4 floats -> 16B alignment holds)
  float* ws     = (float*)d_ws;
  float* agg    = ws;                                      // 50000*512 (reused at 256)
  float* tbuf   = agg    + (size_t)N_NODES * D_IN;         // 50000*256
  float* hA     = tbuf   + (size_t)N_NODES * DIM_H;        // 50000*256
  float* hB     = hA     + (size_t)N_NODES * DIM_H;        // 50000*256
  float* pooled = hB     + (size_t)N_NODES * DIM_H;        // 64*768
  float* lin1o  = pooled + (size_t)N_GRAPHS * 3 * DIM_H;   // 64*64
  _Float16* hp  = (_Float16*)(lin1o + (size_t)N_GRAPHS * 64);
  _Float16* w1h[3], *w2h[3];
  w1h[0] = hp;                 hp += (size_t)D_IN  * DIM_H;
  w2h[0] = hp;                 hp += (size_t)DIM_H * DIM_H;
  w1h[1] = hp;                 hp += (size_t)DIM_H * DIM_H;
  w2h[1] = hp;                 hp += (size_t)DIM_H * DIM_H;
  w1h[2] = hp;                 hp += (size_t)DIM_H * DIM_H;
  w2h[2] = hp;                 hp += (size_t)DIM_H * DIM_H;
  _Float16* l1h = hp;

  const int TB = 256;
  auto blocks = [](long long n, int tb) { return (unsigned)((n + tb - 1) / tb); };

  // weights f32 -> f16, transposed to N-major for coalesced GEMM B staging
  k_convert_wT<<<blocks((long long)D_IN * DIM_H, TB), TB, 0, stream>>>(W1[0], w1h[0], D_IN, DIM_H);
  k_convert_wT<<<blocks((long long)DIM_H * DIM_H, TB), TB, 0, stream>>>(W2[0], w2h[0], DIM_H, DIM_H);
  for (int l = 1; l < 3; ++l) {
    k_convert_wT<<<blocks((long long)DIM_H * DIM_H, TB), TB, 0, stream>>>(W1[l], w1h[l], DIM_H, DIM_H);
    k_convert_wT<<<blocks((long long)DIM_H * DIM_H, TB), TB, 0, stream>>>(W2[l], w2h[l], DIM_H, DIM_H);
  }
  k_convert_wT<<<blocks(3LL * DIM_H * 64, TB), TB, 0, stream>>>(lin1W, l1h, 3 * DIM_H, 64);

  // pooled accumulates via atomics: zero each call
  k_zero<<<blocks((long long)N_GRAPHS * 3 * DIM_H, TB), TB, 0, stream>>>(pooled, N_GRAPHS * 3 * DIM_H);

  dim3 gemmBig(DIM_H / 64, (N_NODES + 255) / 256);  // (4, 196)
  const long long sc512 = (long long)N_EDGES * (D_IN / 4);
  const long long sc256 = (long long)N_EDGES * (DIM_H / 4);
  const long long poolN = (long long)N_NODES * (DIM_H / 4);

  // ---- layer 1 (input dim 512) ----
  k_copy4<<<blocks((long long)N_NODES * D_IN / 4, TB), TB, 0, stream>>>(x, agg, N_NODES * D_IN / 4);
  k_scatter_add<<<blocks(sc512, TB), TB, 0, stream>>>(x, src, dst, agg, D_IN);
  k_wmma_gemm<<<gemmBig, TB, 0, stream>>>(agg, w1h[0], tbuf, N_NODES, DIM_H, D_IN,
                                          b1[0], ga[0], be[0], mu[0], va[0], 1);
  k_wmma_gemm<<<gemmBig, TB, 0, stream>>>(tbuf, w2h[0], hA, N_NODES, DIM_H, DIM_H,
                                          b2[0], nullptr, nullptr, nullptr, nullptr, 1);
  k_pool<<<blocks(poolN, TB), TB, 0, stream>>>(hA, batch, pooled, 0);

  // ---- layer 2 ----
  k_copy4<<<blocks((long long)N_NODES * DIM_H / 4, TB), TB, 0, stream>>>(hA, agg, N_NODES * DIM_H / 4);
  k_scatter_add<<<blocks(sc256, TB), TB, 0, stream>>>(hA, src, dst, agg, DIM_H);
  k_wmma_gemm<<<gemmBig, TB, 0, stream>>>(agg, w1h[1], tbuf, N_NODES, DIM_H, DIM_H,
                                          b1[1], ga[1], be[1], mu[1], va[1], 1);
  k_wmma_gemm<<<gemmBig, TB, 0, stream>>>(tbuf, w2h[1], hB, N_NODES, DIM_H, DIM_H,
                                          b2[1], nullptr, nullptr, nullptr, nullptr, 1);
  k_pool<<<blocks(poolN, TB), TB, 0, stream>>>(hB, batch, pooled, DIM_H);

  // ---- layer 3 ----
  k_copy4<<<blocks((long long)N_NODES * DIM_H / 4, TB), TB, 0, stream>>>(hB, agg, N_NODES * DIM_H / 4);
  k_scatter_add<<<blocks(sc256, TB), TB, 0, stream>>>(hB, src, dst, agg, DIM_H);
  k_wmma_gemm<<<gemmBig, TB, 0, stream>>>(agg, w1h[2], tbuf, N_NODES, DIM_H, DIM_H,
                                          b1[2], ga[2], be[2], mu[2], va[2], 1);
  k_wmma_gemm<<<gemmBig, TB, 0, stream>>>(tbuf, w2h[2], hA, N_NODES, DIM_H, DIM_H,
                                          b2[2], nullptr, nullptr, nullptr, nullptr, 1);
  k_pool<<<blocks(poolN, TB), TB, 0, stream>>>(hA, batch, pooled, 2 * DIM_H);

  // ---- head: lin1 via WMMA (M=64 handled by guards), then lin2 + log_softmax ----
  dim3 gemmLin1(1, 1);
  k_wmma_gemm<<<gemmLin1, TB, 0, stream>>>(pooled, l1h, lin1o, N_GRAPHS, 64, 3 * DIM_H,
                                           lin1b, nullptr, nullptr, nullptr, nullptr, 1);
  k_head<<<1, 64, 0, stream>>>(lin1o, lin2W, lin2b, (float*)d_out);
}
